// SpatialAttentionModel_12670153523500
// MI455X (gfx1250) — matrix operand order repaired and checked by reading, not compiled, not run on hardware
//
#include <hip/hip_runtime.h>

typedef __attribute__((ext_vector_type(16))) _Float16 v16h;
typedef __attribute__((ext_vector_type(8)))  _Float16 h8;
typedef __attribute__((ext_vector_type(8)))  float    v8f;
typedef __attribute__((ext_vector_type(4)))  float    f4;
typedef __attribute__((ext_vector_type(4)))  int      i32x4;

#define NTOK   31200      // B*T*N tokens
#define NROWP  31216      // padded to 1951 16-row tiles
#define NTILE  1951
#define NN     325
#define BTN    96         // B*T
#define DD     64

#if defined(__has_builtin)
# if __has_builtin(__builtin_amdgcn_global_load_async_to_lds_b128) && \
     __has_builtin(__builtin_amdgcn_s_wait_asynccnt)
#  define HAVE_ASYNC_LDS 1
# endif
#endif
#ifndef HAVE_ASYNC_LDS
# define HAVE_ASYNC_LDS 0
#endif

#if HAVE_ASYNC_LDS
typedef __attribute__((address_space(1))) i32x4 i32x4_g;
typedef __attribute__((address_space(3))) i32x4 i32x4_l;
#endif

__device__ __forceinline__ float gelu_exact(float x) {
  return 0.5f * x * (1.0f + erff(x * 0.70710678118654752f));
}
__device__ __forceinline__ int imin(int a, int b) { return a < b ? a : b; }

__device__ __forceinline__ v8f wmma_f16(v16h a, v16h b, v8f c) {
  return __builtin_amdgcn_wmma_f32_16x16x32_f16(false, a, false, b, (short)0, c,
                                                false, false);
}

// build a 16-wide f16 vector from two contiguous f32x8 runs
__device__ __forceinline__ v16h pack16_from_f32(const float* run0, const float* run1) {
  f4 a0 = *(const f4*)(run0);
  f4 a1 = *(const f4*)(run0 + 4);
  f4 b0 = *(const f4*)(run1);
  f4 b1 = *(const f4*)(run1 + 4);
  v16h r;
  #pragma unroll
  for (int j = 0; j < 4; ++j) {
    r[j]      = (_Float16)a0[j];
    r[4 + j]  = (_Float16)a1[j];
    r[8 + j]  = (_Float16)b0[j];
    r[12 + j] = (_Float16)b1[j];
  }
  return r;
}

// ---------------- Kernel 1: q/k/v = gelu(concat(X,STE) @ W^T + b) ------------
__global__ __launch_bounds__(256) void k1_qkv(
    const float* __restrict__ X, const float* __restrict__ STE,
    const float* __restrict__ W7, const float* __restrict__ b7,
    const float* __restrict__ W8, const float* __restrict__ b8,
    const float* __restrict__ W9, const float* __restrict__ b9,
    _Float16* __restrict__ q, _Float16* __restrict__ k, _Float16* __restrict__ v)
{
  const int wave = threadIdx.x >> 5;
  const int lane = threadIdx.x & 31;
  const int lrow = lane & 15;
  const int hf   = lane >> 4;
  const int tile = blockIdx.x * 8 + wave;
  if (tile >= NTILE) return;
  const int row0 = tile * 16;
  const int tokc = imin(row0 + lrow, NTOK - 1);
  const float* xr = X   + (size_t)tokc * DD;
  const float* sr = STE + (size_t)tokc * DD;

  // A fragments: lane row tokc, K runs {c*32+8*hf..+8} and {+16}
  v16h a[4];
  #pragma unroll
  for (int c = 0; c < 4; ++c) {
    const int kA = c * 32 + 8 * hf;          // c<2 -> X, c>=2 -> STE
    const float* base = (c < 2) ? xr : (sr - DD);
    a[c] = pack16_from_f32(base + kA, base + kA + 16);
  }

  const float* Ws[3] = {W7, W8, W9};
  const float* Bs[3] = {b7, b8, b9};
  _Float16*    Os[3] = {q, k, v};
  #pragma unroll
  for (int w = 0; w < 3; ++w) {
    const float sc = (w == 0) ? 0.35355339059327373f : 1.0f; // fold 1/sqrt(d) into q
    #pragma unroll
    for (int ot = 0; ot < 4; ++ot) {
      const int ocol = ot * 16 + lrow;
      v8f acc = {};
      #pragma unroll
      for (int c = 0; c < 4; ++c) {
        const float* wr = Ws[w] + (size_t)ocol * 128 + c * 32 + 8 * hf;
        v16h bf = pack16_from_f32(wr, wr + 16);
        acc = wmma_f16(a[c], bf, acc);
      }
      const float bias = Bs[w][ocol];
      #pragma unroll
      for (int r = 0; r < 8; ++r) {
        const int rt = row0 + r + 8 * hf;
        Os[w][(size_t)rt * DD + ocol] = (_Float16)(gelu_exact(acc[r] + bias) * sc);
      }
    }
  }
}

// ---------------- Kernel 2: fused masked attention, one head per wave --------
__global__ __launch_bounds__(256) void k2_attn(
    const _Float16* __restrict__ q, const _Float16* __restrict__ kk,
    const _Float16* __restrict__ v, const int* __restrict__ adj,
    _Float16* __restrict__ ao)
{
  __shared__ __align__(16) _Float16 sp[8][16][352];   // scores -> P (in place)
  __shared__ __align__(16) _Float16 vstage[8][352][8];// per-head V slice
  __shared__ __align__(16) _Float16 maskl[16][336];   // adj tile, shared by heads
  __shared__ float sinv[8][16];

  const int h    = threadIdx.x >> 5;    // wave == head
  const int lane = threadIdx.x & 31;
  const int lrow = lane & 15;
  const int hf   = lane >> 4;
  const int bt    = blockIdx.x / 21;
  const int ntile = blockIdx.x % 21;
  const int n0    = ntile * 16;
  const size_t base = (size_t)bt * NN;

  // ---- stage V_h[352][8] into LDS (async copy; ASYNCcnt tracked) ----
  #pragma unroll
  for (int it = 0; it < 11; ++it) {
    const int row = it * 32 + lane;
    const int rc  = imin(row, 335);
    const _Float16* src = v + (base + rc) * DD + h * 8;     // 16B
    _Float16* dst = &vstage[h][row][0];                     // 16B aligned
#if HAVE_ASYNC_LDS
    __builtin_amdgcn_global_load_async_to_lds_b128(
        (i32x4_g*)(i32x4*)src, (i32x4_l*)(i32x4*)dst, 0, 0);
#else
    *(h8*)dst = *(const h8*)src;
#endif
  }

  // ---- stage adjacency tile once per block (all 256 threads) ----
  for (int i = 0; i < 21; ++i) {
    const int idx = threadIdx.x + i * 256;  // covers 16*336 exactly
    const int r = idx / 336, m = idx % 336;
    const int n = n0 + r;
    _Float16 mk = (_Float16)0.0f;
    if (m < NN && n < NN) mk = (adj[n * NN + m] > 0) ? (_Float16)1.0f : (_Float16)0.0f;
    maskl[r][m] = mk;
  }
  __syncthreads();

  // ---- Q fragment: K = 8 valid of 32 ----
  v16h qa = {};
  if (hf == 0) {
    h8 qv = *(const h8*)(q + (base + n0 + lrow) * DD + h * 8);
    #pragma unroll
    for (int j = 0; j < 8; ++j) qa[j] = qv[j];
  }

  // ---- scores, masked, f16 into LDS ----
  for (int mt = 0; mt < 21; ++mt) {
    const int m0 = mt * 16;
    v16h kb = {};
    if (hf == 0) {
      const int mc = imin(m0 + lrow, 335);
      h8 kv = *(const h8*)(kk + (base + mc) * DD + h * 8);
      #pragma unroll
      for (int j = 0; j < 8; ++j) kb[j] = kv[j];
    }
    v8f s = {};
    s = wmma_f16(qa, kb, s);
    const int m = m0 + lrow;
    #pragma unroll
    for (int r = 0; r < 8; ++r) {
      const int nl = r + 8 * hf;
      const bool ok = (float)maskl[nl][m] != 0.0f;
      sp[h][nl][m] = ok ? (_Float16)s[r] : (_Float16)(-65504.0f);
    }
  }
  // zero K-padding columns [336,352) for P@V
  #pragma unroll
  for (int j = 0; j < 8; ++j) sp[h][lrow][336 + hf * 8 + j] = (_Float16)0.0f;

  // ---- softmax over 336 cols; lane pair (lrow, lrow+16) splits the row ----
  const int c0 = hf * 168;
  float mx = -3.0e38f;
  for (int m2 = 0; m2 < 168; m2 += 8) {
    h8 sv = *(const h8*)&sp[h][lrow][c0 + m2];
    #pragma unroll
    for (int j = 0; j < 8; ++j) mx = fmaxf(mx, (float)sv[j]);
  }
  mx = fmaxf(mx, __shfl_xor(mx, 16, 32));
  float sum = 0.0f;
  for (int m2 = 0; m2 < 168; m2 += 8) {
    h8 sv = *(const h8*)&sp[h][lrow][c0 + m2];
    h8 pv;
    #pragma unroll
    for (int j = 0; j < 8; ++j) {
      float p = __expf((float)sv[j] - mx);
      sum += p;
      pv[j] = (_Float16)p;
    }
    *(h8*)&sp[h][lrow][c0 + m2] = pv;     // unnormalized P
  }
  sum += __shfl_xor(sum, 16, 32);
  if (hf == 0) sinv[h][lrow] = 1.0f / sum;

#if HAVE_ASYNC_LDS
  __builtin_amdgcn_s_wait_asynccnt(0);    // V slice resident in LDS
#endif

  // ---- out_h = P @ V_h : K = 352 over 11 chunks of 32 ----
  const int cc = lrow;
  const int vc = cc & 7;
  v8f acc = {};
  for (int c = 0; c < 11; ++c) {
    v16h pa, vb;
    h8 p0 = *(const h8*)&sp[h][lrow][c * 32 + 8 * hf];
    h8 p1 = *(const h8*)&sp[h][lrow][c * 32 + 16 + 8 * hf];
    #pragma unroll
    for (int j = 0; j < 8; ++j) { pa[j] = p0[j]; pa[8 + j] = p1[j]; }
    #pragma unroll
    for (int j = 0; j < 16; ++j) {
      const int m = c * 32 + 8 * hf + j + (j >= 8 ? 8 : 0);
      vb[j] = (cc < 8) ? vstage[h][m][vc] : (_Float16)0.0f;
    }
    acc = wmma_f16(pa, vb, acc);
  }
  #pragma unroll
  for (int r = 0; r < 8; ++r) {
    const int nl = r + 8 * hf;
    const int n  = n0 + nl;
    if (cc < 8 && n < NN)
      ao[(base + n) * DD + h * 8 + cc] = (_Float16)(acc[r] * sinv[h][nl]);
  }
}

// ---------------- Kernel 3: out = gelu(ao @ W10^T + b10) @ W11^T + b11 -------
__global__ __launch_bounds__(256) void k3_mlp(
    const _Float16* __restrict__ ao,
    const float* __restrict__ W10, const float* __restrict__ b10,
    const float* __restrict__ W11, const float* __restrict__ b11,
    float* __restrict__ out)
{
  __shared__ __align__(16) _Float16 Hl[8][16][64];   // per-wave hidden tile
  const int wave = threadIdx.x >> 5;
  const int lane = threadIdx.x & 31;
  const int lrow = lane & 15;
  const int hf   = lane >> 4;
  const int tile = blockIdx.x * 8 + wave;
  if (tile >= NTILE) return;                // per-wave LDS region, no barriers
  const int row0 = tile * 16;
  const _Float16* ar = ao + (size_t)(row0 + lrow) * DD;

  __builtin_prefetch(W11 + (size_t)lrow * DD, 0, 1);

  v16h a[2];
  #pragma unroll
  for (int c = 0; c < 2; ++c) {
    h8 a0 = *(const h8*)(ar + c * 32 + 8 * hf);
    h8 a1 = *(const h8*)(ar + c * 32 + 16 + 8 * hf);
    #pragma unroll
    for (int j = 0; j < 8; ++j) { a[c][j] = a0[j]; a[c][8 + j] = a1[j]; }
  }
  #pragma unroll
  for (int ot = 0; ot < 4; ++ot) {
    const int ocol = ot * 16 + lrow;
    v8f acc = {};
    #pragma unroll
    for (int c = 0; c < 2; ++c) {
      const float* wr = W10 + (size_t)ocol * DD + c * 32 + 8 * hf;
      v16h bf = pack16_from_f32(wr, wr + 16);
      acc = wmma_f16(a[c], bf, acc);
    }
    const float bias = b10[ocol];
    #pragma unroll
    for (int r = 0; r < 8; ++r)
      Hl[wave][r + 8 * hf][ocol] = (_Float16)gelu_exact(acc[r] + bias);
  }

  v16h ha[2];
  #pragma unroll
  for (int c = 0; c < 2; ++c) {
    h8 a0 = *(const h8*)&Hl[wave][lrow][c * 32 + 8 * hf];
    h8 a1 = *(const h8*)&Hl[wave][lrow][c * 32 + 16 + 8 * hf];
    #pragma unroll
    for (int j = 0; j < 8; ++j) { ha[c][j] = a0[j]; ha[c][8 + j] = a1[j]; }
  }
  #pragma unroll
  for (int ot = 0; ot < 4; ++ot) {
    const int ocol = ot * 16 + lrow;
    v8f acc = {};
    #pragma unroll
    for (int c = 0; c < 2; ++c) {
      const float* wr = W11 + (size_t)ocol * DD + c * 32 + 8 * hf;
      v16h bf = pack16_from_f32(wr, wr + 16);
      acc = wmma_f16(ha[c], bf, acc);
    }
    const float bias = b11[ocol];
    #pragma unroll
    for (int r = 0; r < 8; ++r) {
      const int tok = row0 + r + 8 * hf;
      if (tok < NTOK) out[(size_t)tok * DD + ocol] = acc[r] + bias;
    }
  }
}

extern "C" void kernel_launch(void* const* d_in, const int* in_sizes, int n_in,
                              void* d_out, int out_size, void* d_ws, size_t ws_size,
                              hipStream_t stream) {
  const float* X   = (const float*)d_in[0];
  const float* STE = (const float*)d_in[1];
  const int*   adj = (const int*)d_in[2];
  const float* W7  = (const float*)d_in[3];
  const float* b7  = (const float*)d_in[4];
  const float* W8  = (const float*)d_in[5];
  const float* b8  = (const float*)d_in[6];
  const float* W9  = (const float*)d_in[7];
  const float* b9  = (const float*)d_in[8];
  const float* W10 = (const float*)d_in[9];
  const float* b10 = (const float*)d_in[10];
  const float* W11 = (const float*)d_in[11];
  const float* b11 = (const float*)d_in[12];
  float* out = (float*)d_out;

  // workspace: q,k,v,attnOut as f16, padded to 31216 rows (≈16 MB, L2-resident)
  _Float16* q  = (_Float16*)d_ws;
  _Float16* k  = q + (size_t)NROWP * DD;
  _Float16* v  = k + (size_t)NROWP * DD;
  _Float16* ao = v + (size_t)NROWP * DD;

  k1_qkv<<<244, 256, 0, stream>>>(X, STE, W7, b7, W8, b8, W9, b9, q, k, v);
  k2_attn<<<BTN * 21, 256, 0, stream>>>(q, k, v, adj, ao);
  k3_mlp<<<244, 256, 0, stream>>>(ao, W10, b10, W11, b11, out);
}
